// GATEncoder_37263136260511
// MI455X (gfx1250) — matrix operand order repaired and checked by reading, not compiled, not run on hardware
//
#include <hip/hip_runtime.h>
#include <hip/hip_bf16.h>

typedef _Float16 v16h __attribute__((ext_vector_type(16)));
typedef _Float16 v8h  __attribute__((ext_vector_type(8)));
typedef float    v8f  __attribute__((ext_vector_type(8)));
typedef float    v4f  __attribute__((ext_vector_type(4)));
typedef unsigned v4u  __attribute__((ext_vector_type(4)));
typedef int      v4i  __attribute__((ext_vector_type(4)));

#define NEG_SLOPE 0.2f
#define LN_EPS    1e-5f

// padded transposed-W layout: Wt[n*136 + k], 128 rows (n), 136 halfs per row
#define WT_STRIDE 136
#define WT_HALFS  (128 * WT_STRIDE)          // 17408 halfs
#define WT_BYTES  (WT_HALFS * 2)             // 34816 bytes
#define WT_CHUNKS (WT_BYTES / 16)            // 2176 x 16B

#define AS1 __attribute__((address_space(1)))
#define AS3 __attribute__((address_space(3)))

#if __has_builtin(__builtin_amdgcn_global_load_async_to_lds_b128) && \
    __has_builtin(__builtin_amdgcn_s_wait_asynccnt)
#define USE_ASYNC_LDS 1
#endif

// ---------- monotone float<->uint key for atomic segment-max ----------
__device__ __forceinline__ unsigned fkey(float f) {
    unsigned u = __float_as_uint(f);
    return (u >> 31) ? ~u : (u | 0x80000000u);
}
__device__ __forceinline__ float fdekey(unsigned k) {
    return __uint_as_float((k >> 31) ? (k & 0x7fffffffu) : ~k);
}

// ---------- W (f32 128x128, [k][n]) -> f16 transposed+padded Wt[n][k] ----------
__global__ void gat_convert_w(const float* __restrict__ W, _Float16* __restrict__ Wt) {
    int i = blockIdx.x * blockDim.x + threadIdx.x;
    if (i >= WT_HALFS) return;
    int n = i / WT_STRIDE, k = i % WT_STRIDE;
    Wt[i] = (k < 128) ? (_Float16)W[k * 128 + n] : (_Float16)0.0f;
}

// ---------- WMMA GEMM: H[N,128] = X[N,128] @ W[128,128] ----------
// 256 threads = 8 waves; Wt cached in LDS; each wave computes a 16x128 stripe.
__global__ void gat_gemm_xw(const float* __restrict__ X, const _Float16* __restrict__ Wt,
                            float* __restrict__ Hout, int nrows) {
    __shared__ _Float16 Wl[WT_HALFS];
    const int tid = threadIdx.x;

    // ---- stage Wt into LDS (async-to-LDS on CDNA5, b128 chunks) ----
#ifdef USE_ASYNC_LDS
    {
        AS1 v4i* g = (AS1 v4i*)Wt;
        AS3 v4i* l = (AS3 v4i*)Wl;
        for (int i = tid; i < WT_CHUNKS; i += 256)
            __builtin_amdgcn_global_load_async_to_lds_b128(g + i, l + i, 0, 0);
        __builtin_amdgcn_s_wait_asynccnt(0);
    }
#else
    {
        const v4u* gs = (const v4u*)Wt;
        v4u* ls = (v4u*)Wl;
        for (int i = tid; i < WT_CHUNKS; i += 256) ls[i] = gs[i];
    }
#endif
    __syncthreads();

    const int wave = tid >> 5;
    const int lane = tid & 31;
    const int half = (lane < 16) ? 0 : 1;
    const int row0 = blockIdx.x * 128 + wave * 16;

    int rowm = row0 + (lane & 15);
    if (rowm >= nrows) rowm = nrows - 1;            // clamp: keeps EXEC all-ones for WMMA
    const v4f* arow4 = (const v4f*)(X + (size_t)rowm * 128);

    const int ncol = lane & 15;
    v8f acc[8] = {};

#pragma unroll
    for (int kc = 0; kc < 4; ++kc) {
        // ---- A fragment (16x32 f16): lanes 0-15 K base 0 / v4-7 base 16; lanes 16-31 +8 ----
        v4f lo0 = arow4[kc * 8 + half * 2];
        v4f lo1 = arow4[kc * 8 + half * 2 + 1];
        v4f hi0 = arow4[kc * 8 + 4 + half * 2];
        v4f hi1 = arow4[kc * 8 + 4 + half * 2 + 1];
        v16h a;
#pragma unroll
        for (int j = 0; j < 4; ++j) {
            a[j]      = (_Float16)lo0[j];
            a[4 + j]  = (_Float16)lo1[j];
            a[8 + j]  = (_Float16)hi0[j];
            a[12 + j] = (_Float16)hi1[j];
        }
#pragma unroll
        for (int nt = 0; nt < 8; ++nt) {
            // ---- B fragment (32x16 f16): lanes 0-15 K base 0 / v4-7 base 16; lanes 16-31 +8 ----
            const _Float16* bp = Wl + (nt * 16 + ncol) * WT_STRIDE + kc * 32 + half * 8;
            v8h b0 = *(const v8h*)bp;           // ds_load_b128
            v8h b1 = *(const v8h*)(bp + 16);    // ds_load_b128
            v16h b;
#pragma unroll
            for (int j = 0; j < 8; ++j) { b[j] = b0[j]; b[8 + j] = b1[j]; }
            acc[nt] = __builtin_amdgcn_wmma_f32_16x16x32_f16(
                false, a, false, b, (short)0, acc[nt], false, false);
        }
    }

    // D layout: VGPR r -> row M = r + 8*half, col N = lane&15
#pragma unroll
    for (int nt = 0; nt < 8; ++nt) {
        const int n = nt * 16 + ncol;
#pragma unroll
        for (int r = 0; r < 8; ++r) {
            int mr = row0 + r + half * 8;
            if (mr < nrows) Hout[(size_t)mr * 128 + n] = acc[nt][r];
        }
    }
}

// ---------- per-(node,head) attention logits ----------
__global__ void gat_node_logits(const float* __restrict__ hfeat,
                                const float* __restrict__ asrc, const float* __restrict__ adst,
                                float* __restrict__ als, float* __restrict__ ald,
                                int N, int H, int C) {
    int t = blockIdx.x * blockDim.x + threadIdx.x;
    if (t >= N * H) return;
    int n = t / H, hh = t % H;
    const float* hp = hfeat + (size_t)n * 128 + hh * C;
    float s1 = 0.f, s2 = 0.f;
    for (int c = 0; c < C; ++c) {
        float v = hp[c];
        s1 += v * asrc[hh * C + c];
        s2 += v * adst[hh * C + c];
    }
    als[t] = s1;   // layout [N,H]
    ald[t] = s2;
}

// ---------- edge pass 1: e = leaky_relu(als[src]+ald[dst]); segment-max via keyed atomicMax ----------
__global__ void gat_edge_logits(const int* __restrict__ ei, int E, int N,
                                const float* __restrict__ als, const float* __restrict__ ald,
                                float* __restrict__ ebuf, unsigned* __restrict__ mkey, int H) {
    long long t = (long long)blockIdx.x * blockDim.x + threadIdx.x;
    long long total = (long long)(E + N) * H;
    if (t >= total) return;
    int edge = (int)(t / H), hh = (int)(t % H);
    int src, dst;
    if (edge < E) { src = ei[edge]; dst = ei[E + edge]; }
    else          { src = dst = edge - E; }                 // self-loop
    float e = als[(size_t)src * H + hh] + ald[(size_t)dst * H + hh];
    e = (e > 0.f) ? e : NEG_SLOPE * e;
    ebuf[(size_t)edge * H + hh] = e;
    atomicMax(&mkey[(size_t)dst * H + hh], fkey(e));
}

// ---------- edge pass 2: one wave per edge; numerator & denominator scatter ----------
__global__ void gat_edge_scatter(const int* __restrict__ ei, int E, int N,
                                 const float* __restrict__ ebuf, const unsigned* __restrict__ mkey,
                                 float* __restrict__ ssum, float* __restrict__ accum,
                                 const float* __restrict__ hfeat, int H, int logC) {
    int wid  = (int)(((long long)blockIdx.x * blockDim.x + threadIdx.x) >> 5);
    int lane = threadIdx.x & 31;
    if (wid >= E + N) return;                                // uniform per wave (wave32)
    int src, dst;
    if (wid < E) { src = ei[wid]; dst = ei[E + wid]; }
    else         { src = dst = wid - E; }
    const float* hs = hfeat + (size_t)src * 128;
    float*       ad = accum + (size_t)dst * 128;
    __builtin_prefetch(hs, 0, 1);                            // global_prefetch_b8
#pragma unroll
    for (int j = 0; j < 4; ++j) {
        int ch   = lane + 32 * j;
        int head = ch >> logC;
        float e  = ebuf[(size_t)wid * H + head];
        float mm = fdekey(mkey[(size_t)dst * H + head]);
        float ex = __expf(e - mm);
        atomicAdd(&ad[ch], ex * hs[ch]);                     // global_atomic_add_f32
    }
    if (lane < H) {
        float e  = ebuf[(size_t)wid * H + lane];
        float mm = fdekey(mkey[(size_t)dst * H + lane]);
        atomicAdd(&ssum[(size_t)dst * H + lane], __expf(e - mm));
    }
}

// ---------- node finalize: divide, +bias, LayerNorm, optional ELU (one wave per node) ----------
__global__ void gat_node_finalize(const float* __restrict__ accum, const float* __restrict__ ssum,
                                  const float* __restrict__ bias, const float* __restrict__ g,
                                  const float* __restrict__ be, float* __restrict__ out,
                                  int N, int H, int logC, int doElu) {
    int node = (int)(((long long)blockIdx.x * blockDim.x + threadIdx.x) >> 5);
    int lane = threadIdx.x & 31;
    if (node >= N) return;
    float v[4];
#pragma unroll
    for (int j = 0; j < 4; ++j) {
        int ch   = lane + 32 * j;
        int head = ch >> logC;
        v[j] = accum[(size_t)node * 128 + ch] / ssum[(size_t)node * H + head] + bias[ch];
    }
    float s = v[0] + v[1] + v[2] + v[3];
#pragma unroll
    for (int off = 16; off; off >>= 1) s += __shfl_xor(s, off, 32);
    float mu = s * (1.f / 128.f);
    float q = 0.f;
#pragma unroll
    for (int j = 0; j < 4; ++j) { float d = v[j] - mu; q += d * d; }
#pragma unroll
    for (int off = 16; off; off >>= 1) q += __shfl_xor(q, off, 32);
    float r = rsqrtf(q * (1.f / 128.f) + LN_EPS);
#pragma unroll
    for (int j = 0; j < 4; ++j) {
        int ch = lane + 32 * j;
        float y = (v[j] - mu) * r * g[ch] + be[ch];
        if (doElu) y = (y > 0.f) ? y : (__expf(y) - 1.f);
        out[(size_t)node * 128 + ch] = y;
    }
}

extern "C" void kernel_launch(void* const* d_in, const int* in_sizes, int n_in,
                              void* d_out, int out_size, void* d_ws, size_t ws_size,
                              hipStream_t stream) {
    const int N  = in_sizes[0] / 128;
    const int E  = in_sizes[1] / 2;
    const int Et = E + N;

    const float* x  = (const float*)d_in[0];
    const int*   ei = (const int*)d_in[1];

    // workspace carve-up (floats)
    float* ws      = (float*)d_ws;
    float* hbuf    = ws;                               // N*128
    float* accum   = hbuf  + (size_t)N * 128;          // N*128
    float* xbuf    = accum + (size_t)N * 128;          // N*128
    float* als     = xbuf  + (size_t)N * 128;          // N*8
    float* ald     = als   + (size_t)N * 8;            // N*8
    unsigned* mkey = (unsigned*)(ald + (size_t)N * 8); // N*8
    float* ssum    = (float*)(mkey + (size_t)N * 8);   // N*8
    float* ebuf    = ssum  + (size_t)N * 8;            // Et*8
    _Float16* Wt   = (_Float16*)(ebuf + (size_t)Et * 8); // WT_HALFS

    auto run_layer = [&](const float* inF, const float* Wp, const float* asrc,
                         const float* adst, const float* bp, const float* gp,
                         const float* bep, int Hh, int Cc, int logC, int doElu, float* outF) {
        gat_convert_w<<<(WT_HALFS + 255) / 256, 256, 0, stream>>>(Wp, Wt);
        gat_gemm_xw<<<(N + 127) / 128, 256, 0, stream>>>(inF, Wt, hbuf, N);
        gat_node_logits<<<(N * Hh + 255) / 256, 256, 0, stream>>>(hbuf, asrc, adst, als, ald, N, Hh, Cc);
        (void)hipMemsetAsync(accum, 0, (size_t)N * 128 * sizeof(float), stream);
        (void)hipMemsetAsync(ssum,  0, (size_t)N * Hh * sizeof(float), stream);
        (void)hipMemsetAsync(mkey,  0, (size_t)N * Hh * sizeof(unsigned), stream);
        long long elThreads = (long long)Et * Hh;
        gat_edge_logits<<<(unsigned)((elThreads + 255) / 256), 256, 0, stream>>>(
            ei, E, N, als, ald, ebuf, mkey, Hh);
        gat_edge_scatter<<<(Et + 7) / 8, 256, 0, stream>>>(
            ei, E, N, ebuf, mkey, ssum, accum, hbuf, Hh, logC);
        gat_node_finalize<<<(N + 7) / 8, 256, 0, stream>>>(
            accum, ssum, bp, gp, bep, outF, N, Hh, logC, doElu);
    };

    const float* W0 = (const float*)d_in[2];
    const float* a_src0 = (const float*)d_in[3];
    const float* a_dst0 = (const float*)d_in[4];
    const float* b0 = (const float*)d_in[5];
    const float* g0 = (const float*)d_in[6];
    const float* be0 = (const float*)d_in[7];
    const float* W1 = (const float*)d_in[8];
    const float* a_src1 = (const float*)d_in[9];
    const float* a_dst1 = (const float*)d_in[10];
    const float* b1 = (const float*)d_in[11];
    const float* g1 = (const float*)d_in[12];
    const float* be1 = (const float*)d_in[13];
    const float* W2 = (const float*)d_in[14];
    const float* a_src2 = (const float*)d_in[15];
    const float* a_dst2 = (const float*)d_in[16];
    const float* b2 = (const float*)d_in[17];
    const float* g2 = (const float*)d_in[18];
    const float* be2 = (const float*)d_in[19];

    run_layer(x,    W0, a_src0, a_dst0, b0, g0, be0, 8, 16,  4, 1, xbuf);
    run_layer(xbuf, W1, a_src1, a_dst1, b1, g1, be1, 8, 16,  4, 1, xbuf);
    run_layer(xbuf, W2, a_src2, a_dst2, b2, g2, be2, 1, 128, 7, 0, (float*)d_out);
}
